// Encoder_32435593020208
// MI455X (gfx1250) — matrix operand order repaired and checked by reading, not compiled, not run on hardware
//
#include <hip/hip_runtime.h>
#include <hip/hip_bf16.h>

// Problem constants (match reference)
#define VOCAB   32000
#define EMBED   1024
#define HIDDEN  1024
#define NLAYERS 2
#define BATCH   128
#define SEQ     256
#define G3      (3*HIDDEN)      // 3072
#define SB      (SEQ*BATCH)     // 32768

typedef __bf16 bf16;
typedef __attribute__((ext_vector_type(8)))  bf16  bf16x8;
typedef __attribute__((ext_vector_type(16))) bf16  bf16x16;
typedef __attribute__((ext_vector_type(8)))  float floatx8;

union ABFrag { bf16x16 v; bf16x8 h[2]; };

__device__ __forceinline__ float sigmoidf_(float x) {
    return 1.0f / (1.0f + __expf(-x));
}

// ---------------------------------------------------------------------------
// fp32 -> bf16 elementwise (weights)
// ---------------------------------------------------------------------------
__global__ void cvt_bf16_kernel(const float* __restrict__ src,
                                bf16* __restrict__ dst, int n) {
    int i = blockIdx.x * blockDim.x + threadIdx.x;
    int stride = gridDim.x * blockDim.x;
    for (; i < n; i += stride) dst[i] = (bf16)src[i];
}

// ---------------------------------------------------------------------------
// Embedding gather with transpose: x0[s][b][e] = embW[source[b][s]][e] (bf16)
// ---------------------------------------------------------------------------
__global__ void embed_kernel(const int* __restrict__ source,
                             const float* __restrict__ embW,
                             bf16* __restrict__ x0) {
    long idx    = (long)blockIdx.x * blockDim.x + threadIdx.x;
    long total  = (long)SB * EMBED;
    long stride = (long)gridDim.x * blockDim.x;
    for (; idx < total; idx += stride) {
        int  e  = (int)(idx & (EMBED - 1));
        long sb = idx >> 10;                 // EMBED = 1024
        int  b  = (int)(sb & (BATCH - 1));
        int  s  = (int)(sb >> 7);            // BATCH = 128
        int tok = source[b * SEQ + s];
        x0[idx] = (bf16)embW[(long)tok * EMBED + e];
    }
}

// ---------------------------------------------------------------------------
// Zero-init recurrent state
// ---------------------------------------------------------------------------
__global__ void zero_h_kernel(float* __restrict__ hf, bf16* __restrict__ hb, int n) {
    int i = blockIdx.x * blockDim.x + threadIdx.x;
    if (i < n) { hf[i] = 0.0f; hb[i] = (bf16)0.0f; }
}

// ---------------------------------------------------------------------------
// Big input-side GEMM: Gi[SB,3H] = X[SB,E](bf16) @ W[3H,E]^T (bf16) + bias
// One wave -> one 16x16 tile; 8 waves/block share the same n-tile (B rows hot
// in L2/WGP$). K loop in steps of 32 using V_WMMA_F32_16X16X32_BF16.
// ---------------------------------------------------------------------------
__global__ void __launch_bounds__(256)
gemm_gi_kernel(const bf16* __restrict__ X,
               const bf16* __restrict__ W,
               const float* __restrict__ bias,
               float* __restrict__ Gi) {
    const int lane  = threadIdx.x & 31;
    const int wave  = threadIdx.x >> 5;
    const int mt    = (blockIdx.x & 255) * 8 + wave;   // 2048 m-tiles / 8
    const int nt    = blockIdx.x >> 8;                 // 192 n-tiles
    const int m     = mt * 16 + (lane & 15);
    const int n     = nt * 16 + (lane & 15);
    const int khalf = lane >> 4;                       // 0 or 1

    const bf16* arow = X + (long)m * EMBED;
    const bf16* brow = W + (long)n * EMBED;

    floatx8 acc = {};
    for (int kk = 0; kk < EMBED; kk += 32) {
        ABFrag a, b;
        // A 16x32 bf16 layout: lanes0-15 K0..15, lanes16-31 K16..31 (matches B)
        a.h[0] = *(const bf16x8*)(arow + kk + khalf * 16);
        a.h[1] = *(const bf16x8*)(arow + kk + khalf * 16 + 8);
        // B 32x16 bf16 layout: lanes0-15 K0..15, lanes16-31 K16..31 (contig in W row)
        b.h[0] = *(const bf16x8*)(brow + kk + khalf * 16);
        b.h[1] = *(const bf16x8*)(brow + kk + khalf * 16 + 8);
        acc = __builtin_amdgcn_wmma_f32_16x16x32_bf16(false, a.v, false, b.v,
                                                      (short)0, acc, false, false);
    }
    const float bn    = bias[n];
    const int   rbase = mt * 16 + khalf * 8;           // C: row = v + 8*(lane>=16)
    for (int v = 0; v < 8; ++v)
        Gi[(long)(rbase + v) * G3 + n] = acc[v] + bn;
}

// ---------------------------------------------------------------------------
// One recurrent timestep, fully fused: per wave one 16x16 (batch x hidden)
// tile with THREE accumulators (r,z,n gates) sharing one A fragment, then
// gate nonlinearity in registers (C fragments of r/z/n align elementwise).
// ---------------------------------------------------------------------------
__global__ void __launch_bounds__(256)
gru_step_kernel(const float* __restrict__ Gi_t,     // [BATCH,3H] (bih included)
                const bf16*  __restrict__ Whh,      // [3H,H] row-major bf16
                const float* __restrict__ bhh,      // [3H]
                const bf16*  __restrict__ hprev_b,  // [BATCH,H]
                const float* __restrict__ hprev_f,  // [BATCH,H]
                bf16*        __restrict__ hnext_b,
                float*       __restrict__ hnext_f,
                bf16*        __restrict__ xout_b,   // next-layer input (or null)
                float*       __restrict__ xout_f,   // d_out ys slice (or null)
                float*       __restrict__ state_out)// d_out state slice (or null)
{
    const int lane  = threadIdx.x & 31;
    const int wave  = threadIdx.x >> 5;
    const int mt    = wave;            // 8 batch tiles
    const int nt    = blockIdx.x;      // 64 hidden tiles
    const int m     = mt * 16 + (lane & 15);
    const int j     = nt * 16 + (lane & 15);
    const int khalf = lane >> 4;

    const bf16* arow = hprev_b + (long)m * HIDDEN;
    const bf16* wr   = Whh + (long)(0 * HIDDEN + j) * HIDDEN;
    const bf16* wz   = Whh + (long)(1 * HIDDEN + j) * HIDDEN;
    const bf16* wn   = Whh + (long)(2 * HIDDEN + j) * HIDDEN;

    floatx8 ar = {}, az = {}, an = {};
    for (int kk = 0; kk < HIDDEN; kk += 32) {
        ABFrag a, b0, b1, b2;
        a.h[0]  = *(const bf16x8*)(arow + kk + khalf * 16);
        a.h[1]  = *(const bf16x8*)(arow + kk + khalf * 16 + 8);
        b0.h[0] = *(const bf16x8*)(wr + kk + khalf * 16);
        b0.h[1] = *(const bf16x8*)(wr + kk + khalf * 16 + 8);
        b1.h[0] = *(const bf16x8*)(wz + kk + khalf * 16);
        b1.h[1] = *(const bf16x8*)(wz + kk + khalf * 16 + 8);
        b2.h[0] = *(const bf16x8*)(wn + kk + khalf * 16);
        b2.h[1] = *(const bf16x8*)(wn + kk + khalf * 16 + 8);
        ar = __builtin_amdgcn_wmma_f32_16x16x32_bf16(false, a.v, false, b0.v,
                                                     (short)0, ar, false, false);
        az = __builtin_amdgcn_wmma_f32_16x16x32_bf16(false, a.v, false, b1.v,
                                                     (short)0, az, false, false);
        an = __builtin_amdgcn_wmma_f32_16x16x32_bf16(false, a.v, false, b2.v,
                                                     (short)0, an, false, false);
    }

    const float bhr = bhh[j];
    const float bhz = bhh[HIDDEN + j];
    const float bhn = bhh[2 * HIDDEN + j];
    const int rbase = mt * 16 + khalf * 8;

    for (int v = 0; v < 8; ++v) {
        const int b = rbase + v;
        const float* gi = Gi_t + (long)b * G3;
        float hr = ar[v] + bhr;
        float hz = az[v] + bhz;
        float hn = an[v] + bhn;
        float r  = sigmoidf_(gi[j] + hr);
        float z  = sigmoidf_(gi[HIDDEN + j] + hz);
        float nn = tanhf(gi[2 * HIDDEN + j] + r * hn);
        float hp = hprev_f[(long)b * HIDDEN + j];
        float hnew = (1.0f - z) * nn + z * hp;
        hnext_f[(long)b * HIDDEN + j] = hnew;
        hnext_b[(long)b * HIDDEN + j] = (bf16)hnew;
        if (xout_b)    xout_b[(long)b * HIDDEN + j]    = (bf16)hnew;
        if (xout_f)    xout_f[(long)b * HIDDEN + j]    = hnew;
        if (state_out) state_out[(long)b * HIDDEN + j] = hnew;
    }
}

// ---------------------------------------------------------------------------
extern "C" void kernel_launch(void* const* d_in, const int* in_sizes, int n_in,
                              void* d_out, int out_size, void* d_ws, size_t ws_size,
                              hipStream_t stream) {
    (void)in_sizes; (void)n_in; (void)out_size; (void)ws_size;

    const int*   source = (const int*)  d_in[0];
    const float* embW   = (const float*)d_in[1];
    const float* W_ih   = (const float*)d_in[2];
    const float* W_hh   = (const float*)d_in[3];
    const float* b_ih   = (const float*)d_in[4];
    const float* b_hh   = (const float*)d_in[5];
    float*       out    = (float*)d_out;

    // Workspace carve-out (256-B aligned): ~538 MiB total
    char*  ws  = (char*)d_ws;
    size_t off = 0;
    auto carve = [&](size_t bytes) -> char* {
        char* p = ws + off;
        off += (bytes + 255) & ~(size_t)255;
        return p;
    };
    bf16*  Wih_b = (bf16*)carve(sizeof(bf16) * (size_t)NLAYERS * G3 * EMBED);
    bf16*  Whh_b = (bf16*)carve(sizeof(bf16) * (size_t)NLAYERS * G3 * HIDDEN);
    bf16*  x0_b  = (bf16*)carve(sizeof(bf16) * (size_t)SB * EMBED);
    bf16*  x1_b  = (bf16*)carve(sizeof(bf16) * (size_t)SB * HIDDEN);
    float* Gi    = (float*)carve(sizeof(float) * (size_t)SB * G3);
    float* hf[2]; bf16* hb[2];
    hf[0] = (float*)carve(sizeof(float) * BATCH * HIDDEN);
    hf[1] = (float*)carve(sizeof(float) * BATCH * HIDDEN);
    hb[0] = (bf16*) carve(sizeof(bf16)  * BATCH * HIDDEN);
    hb[1] = (bf16*) carve(sizeof(bf16)  * BATCH * HIDDEN);

    // 1) Weights -> bf16 (kept in reference row-major [3H,K] layout: B-fragment
    //    rows are K-contiguous, no transpose instruction needed).
    const int nW = NLAYERS * G3 * EMBED;   // == NLAYERS*G3*HIDDEN
    cvt_bf16_kernel<<<2048, 256, 0, stream>>>(W_ih, Wih_b, nW);
    cvt_bf16_kernel<<<2048, 256, 0, stream>>>(W_hh, Whh_b, nW);

    // 2) Embedding gather ([S,B,E], bf16)
    embed_kernel<<<4096, 256, 0, stream>>>(source, embW, x0_b);

    const bf16* xin = x0_b;
    for (int l = 0; l < NLAYERS; ++l) {
        // 3) Hoisted input-side GEMM for all timesteps of this layer
        gemm_gi_kernel<<<256 * (G3 / 16), 256, 0, stream>>>(
            xin, Wih_b + (size_t)l * G3 * EMBED, b_ih + (size_t)l * G3, Gi);

        // 4) h0 = 0
        zero_h_kernel<<<(BATCH * HIDDEN + 255) / 256, 256, 0, stream>>>(
            hf[0], hb[0], BATCH * HIDDEN);

        // 5) Sequential scan: one fused WMMA+gate kernel per timestep
        for (int t = 0; t < SEQ; ++t) {
            const int cur = t & 1, nxt = cur ^ 1;
            bf16*  xob = (l == 0) ? x1_b + (size_t)t * BATCH * HIDDEN : nullptr;
            float* xof = (l == NLAYERS - 1) ? out + (size_t)t * BATCH * HIDDEN : nullptr;
            float* st  = (t == SEQ - 1)
                       ? out + (size_t)SB * HIDDEN + (size_t)l * BATCH * HIDDEN
                       : nullptr;
            gru_step_kernel<<<HIDDEN / 16, 256, 0, stream>>>(
                Gi + (size_t)t * BATCH * G3,
                Whh_b + (size_t)l * G3 * HIDDEN,
                b_hh + (size_t)l * G3,
                hb[cur], hf[cur], hb[nxt], hf[nxt],
                xob, xof, st);
        }
        xin = x1_b;
    }
}
